// MultiheadSelfAttentionWithPairwiseBias_15350213115909
// MI455X (gfx1250) — matrix operand order repaired and checked by reading, not compiled, not run on hardware
//
#include <hip/hip_runtime.h>

typedef __attribute__((ext_vector_type(2)))  float          v2f;
typedef __attribute__((ext_vector_type(8)))  float          v8f;
typedef __attribute__((ext_vector_type(4)))  int            v4i;
typedef __attribute__((ext_vector_type(16))) __bf16         v16bf;
typedef __attribute__((ext_vector_type(8)))  unsigned short v8u;

union BFU { v16bf v; v8u p[2]; };

#define B_     16
#define N_     512
#define D_     512
#define H_     8
#define DH_    64
#define QKV_LD 1536   // 3*D

#if defined(__has_builtin)
#  if __has_builtin(__builtin_amdgcn_global_load_async_to_lds_b128)
#    define USE_ASYNC_LDS 1
#  endif
#endif

__device__ __forceinline__ v8f wmma_f32(v2f a, v2f b, v8f c) {
  return __builtin_amdgcn_wmma_f32_16x16x4_f32(false, a, false, b, (short)0, c, false, false);
}
__device__ __forceinline__ v8f wmma_bf(v16bf a, v16bf b, v8f c) {
  return __builtin_amdgcn_wmma_f32_16x16x32_bf16(false, a, false, b, (short)0, c, false, false);
}

// split f32 x into bf16 hi + bf16 lo (x ~= hi + lo, |err| ~ 2^-17 |x|)
__device__ __forceinline__ void split_bf16(float x, unsigned short* ph, unsigned short* pl) {
  unsigned int bx = __float_as_uint(x);
  *ph = (unsigned short)(bx >> 16);
  float hf = __uint_as_float(bx & 0xffff0000u);
  *pl = (unsigned short)(__float_as_uint(x - hf) >> 16);
}

// ---------------------------------------------------------------------------
// Precision-split conversion kernels (pure bandwidth, ~2us total at 23 TB/s)
// ---------------------------------------------------------------------------
__global__ __launch_bounds__(256) void split_kernel(
    const float* __restrict__ src, unsigned short* __restrict__ hi,
    unsigned short* __restrict__ lo, int n) {
  int i = blockIdx.x * 256 + threadIdx.x;
  if (i < n) {
    unsigned short h_, l_;
    split_bf16(src[i], &h_, &l_);
    hi[i] = h_; lo[i] = l_;
  }
}

// src is K x Nc row-major; dst is Nc x K (transposed) so every B fragment
// (one lane's 16 contiguous K values, this code's k order) is one 32B load.
__global__ __launch_bounds__(256) void split_t_kernel(
    const float* __restrict__ src, unsigned short* __restrict__ hi,
    unsigned short* __restrict__ lo, int K, int Nc) {
  int i = blockIdx.x * 256 + threadIdx.x;
  if (i < K * Nc) {
    int k = i / Nc, n = i % Nc;
    unsigned short h_, l_;
    split_bf16(src[i], &h_, &l_);
    hi[(size_t)n * K + k] = h_;
    lo[(size_t)n * K + k] = l_;
  }
}

// ---------------------------------------------------------------------------
// Kernel 1: qkv = h @ W_qkv via split-bf16 WMMA 16x16x32 (3 wmma per k-chunk)
// one wave -> 16-row M tile x 64-col group; A fragments reused by 4 N tiles
// A 16-bit frag: VGPR0-3 = K[16h..16h+7], VGPR4-7 = K[16h+8..16h+15] (contig runs)
// ---------------------------------------------------------------------------
__global__ __launch_bounds__(256) void qkv_proj_kernel(
    const unsigned short* __restrict__ h_hi, const unsigned short* __restrict__ h_lo,
    const unsigned short* __restrict__ wt_hi, const unsigned short* __restrict__ wt_lo,
    float* __restrict__ qkv) {
  int wid  = (blockIdx.x << 3) + (threadIdx.x >> 5);
  int lane = threadIdx.x & 31;
  int tm   = wid / 24;          // 512 M tiles
  int tg   = wid % 24;          // 24 groups of 64 cols
  int half = lane >> 4;
  int l16  = lane & 15;

  const unsigned short* ah = h_hi + (size_t)(tm * 16 + l16) * D_;
  const unsigned short* al = h_lo + (size_t)(tm * 16 + l16) * D_;
  const unsigned short* bh[4]; const unsigned short* bl[4];
#pragma unroll
  for (int nt = 0; nt < 4; ++nt) {
    int col = tg * 64 + nt * 16 + l16;     // Wt is [1536][512]
    bh[nt] = wt_hi + (size_t)col * D_;
    bl[nt] = wt_lo + (size_t)col * D_;
  }

  v8f acc0 = {}, acc1 = {}, acc2 = {}, acc3 = {};
  for (int k = 0; k < D_; k += 32) {
    BFU Ah, Al;
    Ah.p[0] = *(const v8u*)(ah + k + 16 * half);
    Ah.p[1] = *(const v8u*)(ah + k + 16 * half + 8);
    Al.p[0] = *(const v8u*)(al + k + 16 * half);
    Al.p[1] = *(const v8u*)(al + k + 16 * half + 8);
    int kb = k + 16 * half;                // per-lane 16 contiguous K values
    v16bf B0h = *(const v16bf*)(const void*)(bh[0] + kb);
    v16bf B1h = *(const v16bf*)(const void*)(bh[1] + kb);
    v16bf B2h = *(const v16bf*)(const void*)(bh[2] + kb);
    v16bf B3h = *(const v16bf*)(const void*)(bh[3] + kb);
    v16bf B0l = *(const v16bf*)(const void*)(bl[0] + kb);
    v16bf B1l = *(const v16bf*)(const void*)(bl[1] + kb);
    v16bf B2l = *(const v16bf*)(const void*)(bl[2] + kb);
    v16bf B3l = *(const v16bf*)(const void*)(bl[3] + kb);
    acc0 = wmma_bf(Ah.v, B0h, acc0); acc0 = wmma_bf(Al.v, B0h, acc0); acc0 = wmma_bf(Ah.v, B0l, acc0);
    acc1 = wmma_bf(Ah.v, B1h, acc1); acc1 = wmma_bf(Al.v, B1h, acc1); acc1 = wmma_bf(Ah.v, B1l, acc1);
    acc2 = wmma_bf(Ah.v, B2h, acc2); acc2 = wmma_bf(Al.v, B2h, acc2); acc2 = wmma_bf(Ah.v, B2l, acc2);
    acc3 = wmma_bf(Ah.v, B3h, acc3); acc3 = wmma_bf(Al.v, B3h, acc3); acc3 = wmma_bf(Ah.v, B3l, acc3);
  }
  float* orow = qkv + (size_t)(tm * 16 + 8 * half) * QKV_LD + tg * 64 + l16;
#pragma unroll
  for (int r = 0; r < 8; ++r) {
    orow[(size_t)r * QKV_LD + 0 ] = acc0[r];
    orow[(size_t)r * QKV_LD + 16] = acc1[r];
    orow[(size_t)r * QKV_LD + 32] = acc2[r];
    orow[(size_t)r * QKV_LD + 48] = acc3[r];
  }
}

// ---------------------------------------------------------------------------
// Kernel 2: fused bias-MLP + flash softmax + PV (exact f32 WMMA path).
// grid = B*N/16 blocks, 256 threads = 8 waves, wave w owns head w.
// K j-tile (16 keys x 512 floats) staged to LDS with async-to-LDS DMA,
// overlapped with the bias MLP; row stride 516 dwords -> conflict-free frags.
// ---------------------------------------------------------------------------
#define KLD 516
__global__ __launch_bounds__(256) void attn_kernel(
    const float* __restrict__ qkv, const float* __restrict__ v4,
    const int* __restrict__ msk, const int* __restrict__ tok,
    const float* __restrict__ W1, const float* __restrict__ b1,
    const float* __restrict__ W2, const float* __restrict__ b2,
    unsigned short* __restrict__ ctx_hi, unsigned short* __restrict__ ctx_lo) {
  __shared__ float w1s[288], b1s[32], w2s[256], b2s[8];
  __shared__ float biasLds[8 * 256];   // [head][i_local][j_local]
  __shared__ float pLds[8 * 256];      // per-wave P tile, C->A relayout
  __shared__ float kLds[16 * KLD];     // 16 keys x 512 (padded) K slab

  int tid = threadIdx.x;
  int b   = blockIdx.x >> 5;
  int it  = blockIdx.x & 31;
  int i0  = it * 16;

  for (int idx = tid; idx < 288; idx += 256) w1s[idx] = W1[idx];
  if (tid < 32) b1s[tid] = b1[tid];
  w2s[tid & 255] = W2[tid & 255];
  if (tid < 8) b2s[tid] = b2[tid];

  int mi = tid >> 4;
  int iq = i0 + mi;
  const float* vip = v4 + (size_t)(b * N_ + iq) * 4;
  float vi0 = vip[0], vi1 = vip[1], vi2 = vip[2], vi3 = vip[3];
  int ti = tok[iq];

  int w = tid >> 5, lane = tid & 31;
  int half = lane >> 4, l16 = lane & 15;

  const float* qrow = qkv + (size_t)(b * N_ + i0 + l16) * QKV_LD + w * DH_;
  v2f qa[16];
#pragma unroll
  for (int t = 0; t < 16; ++t) qa[t] = *(const v2f*)(qrow + 4 * t + 2 * half);

  v8f o0 = {}, o1 = {}, o2 = {}, o3 = {};
  float runm[8], runs[8];
#pragma unroll
  for (int r = 0; r < 8; ++r) { runm[r] = -1e30f; runs[r] = 0.f; }

  __syncthreads();

  for (int jt = 0; jt < 32; ++jt) {
    int j0 = jt * 16;

    // ---- stage K slab (16 keys x 512 f32) to LDS, overlapped with bias ----
    {
      const float* kbase = qkv + (size_t)(b * N_ + j0) * QKV_LD + 512;
#pragma unroll
      for (int c = 0; c < 8; ++c) {
        int id  = (c << 8) + tid;          // 0..2047 b128 chunks
        int row = id >> 7;
        int cc  = (id & 127) << 2;
#ifdef USE_ASYNC_LDS
        __builtin_amdgcn_global_load_async_to_lds_b128(
            (__attribute__((address_space(1))) v4i*)(kbase + (size_t)row * QKV_LD + cc),
            (__attribute__((address_space(3))) v4i*)(&kLds[row * KLD + cc]), 0, 0);
#else
        *(float4*)(&kLds[row * KLD + cc]) =
            *(const float4*)(kbase + (size_t)row * QKV_LD + cc);
#endif
      }
    }

    // ---- pairwise bias MLP: one (i,j) pair per lane, all heads at once ----
    {
      int nj = tid & 15;
      int j  = j0 + nj;
      const float* vjp = v4 + (size_t)(b * N_ + j) * 4;
      float d0 = vi0 - vjp[0], d1 = vi1 - vjp[1], d2 = vi2 - vjp[2], d3 = vi3 - vjp[3];
      float p0 = vi0 * vjp[0], p1 = vi1 * vjp[1], p2 = vi2 * vjp[2], p3 = vi3 * vjp[3];
      float sm = (ti == tok[j]) ? 1.f : 0.f;
      float ob[8];
#pragma unroll
      for (int o = 0; o < 8; ++o) ob[o] = b2s[o];
#pragma unroll
      for (int c = 0; c < 32; ++c) {
        float hh = b1s[c]
          + d0 * w1s[c]       + d1 * w1s[32 + c]  + d2 * w1s[64 + c]  + d3 * w1s[96 + c]
          + p0 * w1s[128 + c] + p1 * w1s[160 + c] + p2 * w1s[192 + c] + p3 * w1s[224 + c]
          + sm * w1s[256 + c];
        hh = fmaxf(hh, 0.f);
#pragma unroll
        for (int o = 0; o < 8; ++o) ob[o] += hh * w2s[(c << 3) + o];
      }
#pragma unroll
      for (int o = 0; o < 8; ++o) biasLds[(o << 8) + tid] = ob[o];
    }
#ifdef USE_ASYNC_LDS
#  if __has_builtin(__builtin_amdgcn_s_wait_asynccnt)
    __builtin_amdgcn_s_wait_asynccnt(0);
#  else
    asm volatile("s_wait_asynccnt 0x0" ::: "memory");
#  endif
#endif
    __syncthreads();

    // ---- S = Q K^T (16 exact f32 WMMAs; K frags from LDS, bank-clean) ----
    const float* krow = &kLds[l16 * KLD + w * DH_];
    v8f s = {};
#pragma unroll
    for (int t = 0; t < 16; ++t) {
      v2f kb = *(const v2f*)(krow + 4 * t + 2 * half);
      s = wmma_f32(qa[t], kb, s);
    }
    int mv = msk[b * N_ + j0 + l16];

    // ---- online softmax per row (rows live in 16-lane halves) ----
#pragma unroll
    for (int r = 0; r < 8; ++r) {
      float val = s[r] * 0.125f + biasLds[(w << 8) + ((r + 8 * half) << 4) + l16];
      val = mv ? val : -1e30f;
      float rm = val;
      rm = fmaxf(rm, __shfl_xor(rm, 1, 32));
      rm = fmaxf(rm, __shfl_xor(rm, 2, 32));
      rm = fmaxf(rm, __shfl_xor(rm, 4, 32));
      rm = fmaxf(rm, __shfl_xor(rm, 8, 32));
      float nm   = fmaxf(runm[r], rm);
      float corr = __expf(runm[r] - nm);
      float p    = __expf(val - nm);
      float ps = p;
      ps += __shfl_xor(ps, 1, 32);
      ps += __shfl_xor(ps, 2, 32);
      ps += __shfl_xor(ps, 4, 32);
      ps += __shfl_xor(ps, 8, 32);
      runs[r] = runs[r] * corr + ps;
      runm[r] = nm;
      o0[r] *= corr; o1[r] *= corr; o2[r] *= corr; o3[r] *= corr;
      pLds[(w << 8) + ((r + 8 * half) << 4) + l16] = p;
    }
    __syncthreads();   // pLds ready; biasLds/kLds free for next jt

    // ---- O += P @ V (C-layout P relayouted via LDS into A fragments) ----
    const float* vrow = qkv + (size_t)(b * N_ + j0) * QKV_LD + 1024 + w * DH_ + l16;
#pragma unroll
    for (int t = 0; t < 4; ++t) {
      v2f pa = *(const v2f*)(&pLds[(w << 8) + (l16 << 4) + 4 * t + 2 * half]);
      const float* vr0 = vrow + (size_t)(4 * t + 2 * half) * QKV_LD;
      const float* vr1 = vr0 + QKV_LD;
      v2f vb0 = { vr0[0],  vr1[0]  };
      v2f vb1 = { vr0[16], vr1[16] };
      v2f vb2 = { vr0[32], vr1[32] };
      v2f vb3 = { vr0[48], vr1[48] };
      o0 = wmma_f32(pa, vb0, o0);
      o1 = wmma_f32(pa, vb1, o1);
      o2 = wmma_f32(pa, vb2, o2);
      o3 = wmma_f32(pa, vb3, o3);
    }
  }

  // normalize; store ctx in split-bf16 directly (feeds the bf16 out-proj)
  size_t cbase = (size_t)(b * N_ + i0 + 8 * half) * D_ + w * DH_ + l16;
#pragma unroll
  for (int r = 0; r < 8; ++r) {
    float inv = 1.0f / runs[r];
    unsigned short hx, lx;
    split_bf16(o0[r] * inv, &hx, &lx); ctx_hi[cbase + (size_t)r * D_ + 0 ] = hx; ctx_lo[cbase + (size_t)r * D_ + 0 ] = lx;
    split_bf16(o1[r] * inv, &hx, &lx); ctx_hi[cbase + (size_t)r * D_ + 16] = hx; ctx_lo[cbase + (size_t)r * D_ + 16] = lx;
    split_bf16(o2[r] * inv, &hx, &lx); ctx_hi[cbase + (size_t)r * D_ + 32] = hx; ctx_lo[cbase + (size_t)r * D_ + 32] = lx;
    split_bf16(o3[r] * inv, &hx, &lx); ctx_hi[cbase + (size_t)r * D_ + 48] = hx; ctx_lo[cbase + (size_t)r * D_ + 48] = lx;
  }
}

// ---------------------------------------------------------------------------
// Kernel 3: out = (ctx @ W_out) * rowmask via split-bf16 WMMA 16x16x32
// ---------------------------------------------------------------------------
__global__ __launch_bounds__(256) void out_proj_kernel(
    const unsigned short* __restrict__ c_hi, const unsigned short* __restrict__ c_lo,
    const unsigned short* __restrict__ wt_hi, const unsigned short* __restrict__ wt_lo,
    const int* __restrict__ msk, float* __restrict__ out) {
  int wid  = (blockIdx.x << 3) + (threadIdx.x >> 5);
  int lane = threadIdx.x & 31;
  int tm   = wid >> 3;          // 512 M tiles
  int tg   = wid & 7;           // 8 groups of 64 cols
  int half = lane >> 4;
  int l16  = lane & 15;

  const unsigned short* ah = c_hi + (size_t)(tm * 16 + l16) * D_;
  const unsigned short* al = c_lo + (size_t)(tm * 16 + l16) * D_;
  const unsigned short* bh[4]; const unsigned short* bl[4];
#pragma unroll
  for (int nt = 0; nt < 4; ++nt) {
    int col = tg * 64 + nt * 16 + l16;     // Wout_t is [512][512]
    bh[nt] = wt_hi + (size_t)col * D_;
    bl[nt] = wt_lo + (size_t)col * D_;
  }

  v8f a0 = {}, a1 = {}, a2 = {}, a3 = {};
  for (int k = 0; k < D_; k += 32) {
    BFU Ah, Al;
    Ah.p[0] = *(const v8u*)(ah + k + 16 * half);
    Ah.p[1] = *(const v8u*)(ah + k + 16 * half + 8);
    Al.p[0] = *(const v8u*)(al + k + 16 * half);
    Al.p[1] = *(const v8u*)(al + k + 16 * half + 8);
    int kb = k + 16 * half;
    v16bf B0h = *(const v16bf*)(const void*)(bh[0] + kb);
    v16bf B1h = *(const v16bf*)(const void*)(bh[1] + kb);
    v16bf B2h = *(const v16bf*)(const void*)(bh[2] + kb);
    v16bf B3h = *(const v16bf*)(const void*)(bh[3] + kb);
    v16bf B0l = *(const v16bf*)(const void*)(bl[0] + kb);
    v16bf B1l = *(const v16bf*)(const void*)(bl[1] + kb);
    v16bf B2l = *(const v16bf*)(const void*)(bl[2] + kb);
    v16bf B3l = *(const v16bf*)(const void*)(bl[3] + kb);
    a0 = wmma_bf(Ah.v, B0h, a0); a0 = wmma_bf(Al.v, B0h, a0); a0 = wmma_bf(Ah.v, B0l, a0);
    a1 = wmma_bf(Ah.v, B1h, a1); a1 = wmma_bf(Al.v, B1h, a1); a1 = wmma_bf(Ah.v, B1l, a1);
    a2 = wmma_bf(Ah.v, B2h, a2); a2 = wmma_bf(Al.v, B2h, a2); a2 = wmma_bf(Ah.v, B2l, a2);
    a3 = wmma_bf(Ah.v, B3h, a3); a3 = wmma_bf(Al.v, B3h, a3); a3 = wmma_bf(Ah.v, B3l, a3);
  }
#pragma unroll
  for (int r = 0; r < 8; ++r) {
    int rowg = tm * 16 + r + 8 * half;
    float keep = msk[rowg] ? 1.0f : 0.0f;
    float* orow = out + (size_t)rowg * D_ + tg * 64 + l16;
    orow[0 ] = a0[r] * keep;
    orow[16] = a1[r] * keep;
    orow[32] = a2[r] * keep;
    orow[48] = a3[r] * keep;
  }
}

// ---------------------------------------------------------------------------
extern "C" void kernel_launch(void* const* d_in, const int* in_sizes, int n_in,
                              void* d_out, int out_size, void* d_ws, size_t ws_size,
                              hipStream_t stream) {
  (void)in_sizes; (void)n_in; (void)out_size; (void)ws_size;
  const float* h    = (const float*)d_in[0];
  const float* v    = (const float*)d_in[1];
  const int*   m    = (const int*)d_in[2];
  const int*   tok  = (const int*)d_in[3];
  const float* Wqkv = (const float*)d_in[4];
  const float* Wout = (const float*)d_in[5];
  const float* W1   = (const float*)d_in[6];
  const float* b1   = (const float*)d_in[7];
  const float* W2   = (const float*)d_in[8];
  const float* b2   = (const float*)d_in[9];
  float* out = (float*)d_out;

  // workspace layout (88.1 MB total)
  char* ws = (char*)d_ws;
  float*          qkv   = (float*)ws;                           // 50,331,648 B
  unsigned short* h_hi  = (unsigned short*)(ws + 50331648ULL);  // 8,388,608 B
  unsigned short* h_lo  = h_hi  + 4194304;                      // 8,388,608 B
  unsigned short* wq_hi = h_lo  + 4194304;                      // 1,572,864 B
  unsigned short* wq_lo = wq_hi + 786432;                       // 1,572,864 B
  unsigned short* wo_hi = wq_lo + 786432;                       //   524,288 B
  unsigned short* wo_lo = wo_hi + 262144;                       //   524,288 B
  unsigned short* cx_hi = wo_lo + 262144;                       // 8,388,608 B
  unsigned short* cx_lo = cx_hi + 4194304;                      // 8,388,608 B

  split_kernel  <<<16384, 256, 0, stream>>>(h, h_hi, h_lo, B_ * N_ * D_);
  split_t_kernel<<< 3072, 256, 0, stream>>>(Wqkv, wq_hi, wq_lo, D_, 3 * D_);
  split_t_kernel<<< 1024, 256, 0, stream>>>(Wout, wo_hi, wo_lo, D_, D_);

  qkv_proj_kernel<<<1536, 256, 0, stream>>>(h_hi, h_lo, wq_hi, wq_lo, qkv);
  attn_kernel    <<< 512, 256, 0, stream>>>(qkv, v, m, tok, W1, b1, W2, b2, cx_hi, cx_lo);
  out_proj_kernel<<< 512, 256, 0, stream>>>(cx_hi, cx_lo, wo_hi, wo_lo, m, out);
}
